// EngramANNInjection_40742059770463
// MI455X (gfx1250) — matrix operand order repaired and checked by reading, not compiled, not run on hardware
//
#include <hip/hip_runtime.h>
#include <hip/hip_bf16.h>
#include <stddef.h>

// ---------------------------------------------------------------------------
// EngramANNInjection for gfx1250 (MI455X).
//   K1  tabprep : table -> bf16 copy + per-row 1/||row||
//   K1b ttrans  : bf16 table -> transposed copy tabT[e][t] (L2-resident)
//   K2  qproj   : q = hidden @ Wq, l2norm -> bf16 qn   (async-to-LDS staging)
//   K3  attn    : flash-style softmax retrieval, all-WMMA, no LDS staging
//   K4  kvgate  : key/value proj + rmsnorm gate (TDM tensor_load_to_lds)
//   K5  conv    : causal depthwise conv K=3 + SiLU
// ---------------------------------------------------------------------------

typedef __attribute__((ext_vector_type(16))) __bf16 v16bf;
typedef __attribute__((ext_vector_type(8)))  float  v8f;

#define HIDDEN   2048
#define ENGRAM   256
#define ENTRIES  32768
#define ROWS     4096            // B*L
#define INV_TEMP 16.0f           // sqrt(ENGRAM)
#define LN_EPS   1e-6f
#define NORM_EPS 1e-8f

// workspace layout (bytes)
#define QN_OFF    0ull                                   // 4096*256 bf16   = 2 MB
#define TAB_OFF   (QN_OFF   + 4096ull*256*2)             // 32768*256 bf16  = 16 MB
#define TABT_OFF  (TAB_OFF  + 32768ull*256*2)            // 256*32768 bf16  = 16 MB
#define INV_OFF   (TABT_OFF + 32768ull*256*2)            // 32768 f32       = 128 KB
#define RET_OFF   (INV_OFF  + 32768ull*4)                // 4096*256 f32    = 4 MB
#define GATED_OFF (RET_OFF  + 4096ull*256*4)             // 4096*2048 f32   = 32 MB

#if __has_builtin(__builtin_amdgcn_global_load_async_to_lds_b128)
#define HAVE_ASYNC_LDS 1
// param 0 per diagnostic: '__attribute__((__vector_size__(4*sizeof(int)))) int __device__ *'
typedef int v4i_vs __attribute__((vector_size(16)));
typedef __attribute__((address_space(1))) v4i_vs* as1_v4i_p;
typedef __attribute__((address_space(3))) v4i_vs* as3_v4i_p;
#endif

#if __has_builtin(__builtin_amdgcn_tensor_load_to_lds) && \
    __has_builtin(__builtin_amdgcn_s_wait_tensorcnt)
#define HAVE_TDM 1
typedef unsigned int v4u_vs __attribute__((vector_size(16)));
typedef int          v8i_vs __attribute__((vector_size(32)));
typedef int          v4i_tdm __attribute__((vector_size(16)));
#endif

__device__ __forceinline__ void wait_async_zero() {
#if __has_builtin(__builtin_amdgcn_s_wait_asynccnt)
  __builtin_amdgcn_s_wait_asynccnt(0);
#else
  asm volatile("s_wait_asynccnt 0x0" ::: "memory");
#endif
}

__device__ __forceinline__ unsigned short bf_bits(float f) {
  union { float f; unsigned u; } v; v.f = f;
  unsigned r = v.u + 0x7FFFu + ((v.u >> 16) & 1u);   // round-to-nearest-even
  return (unsigned short)(r >> 16);
}

__device__ __forceinline__ float block_reduce_256(float v, float* red) {
  const int tid = threadIdx.x;
  red[tid] = v; __syncthreads();
  #pragma unroll
  for (int s = 128; s > 0; s >>= 1) {
    if (tid < s) red[tid] += red[tid + s];
    __syncthreads();
  }
  float r = red[0]; __syncthreads();
  return r;
}

// --------------------------- K1: table prep --------------------------------
__global__ __launch_bounds__(256) void tabprep_kernel(
    const float* __restrict__ table, unsigned short* __restrict__ tab_bf,
    float* __restrict__ invn) {
  __shared__ float red[256];
  const int row = blockIdx.x, tid = threadIdx.x;
  const float x = table[(size_t)row * ENGRAM + tid];
  const float ss = block_reduce_256(x * x, red);
  if (tid == 0) invn[row] = 1.0f / fmaxf(sqrtf(ss), NORM_EPS);
  tab_bf[(size_t)row * ENGRAM + tid] = bf_bits(x);
}

// --------------------------- K1b: table transpose --------------------------
// tabT[e * ENTRIES + t] = tab[t * ENGRAM + e]; coalesced reads and writes
__global__ __launch_bounds__(256) void ttrans_kernel(
    const unsigned short* __restrict__ tab, unsigned short* __restrict__ tabT) {
  const int t0 = (blockIdx.x & 127) * 256;
  const int e0 = (blockIdx.x >> 7) * 16;
  const int tid = threadIdx.x;
  unsigned short v[16];
  __builtin_memcpy(v, tab + (size_t)(t0 + tid) * ENGRAM + e0, 32);
  #pragma unroll
  for (int r = 0; r < 16; ++r)
    tabT[(size_t)(e0 + r) * ENTRIES + t0 + tid] = v[r];
}

// --------------------------- K2: q projection ------------------------------
__global__ __launch_bounds__(256) void qproj_kernel(
    const float* __restrict__ hs, const float* __restrict__ Wq,
    unsigned short* __restrict__ qn) {
  __shared__ __align__(16) float hrow[HIDDEN];
  __shared__ float red[256];
  const int row = blockIdx.x, tid = threadIdx.x;
  const float* h = hs + (size_t)row * HIDDEN;
#if HAVE_ASYNC_LDS
  {
    as1_v4i_p g = (as1_v4i_p)(h);       // 2048 f32 = 512 x v4i
    as3_v4i_p l = (as3_v4i_p)(hrow);
    __builtin_amdgcn_global_load_async_to_lds_b128(g + tid, l + tid, 0, 0);
    __builtin_amdgcn_global_load_async_to_lds_b128(g + 256 + tid,
                                                   l + 256 + tid, 0, 0);
    wait_async_zero();
  }
  __syncthreads();
#else
  for (int i = tid; i < HIDDEN; i += 256) hrow[i] = h[i];
  __syncthreads();
#endif
  float q = 0.0f;
  #pragma unroll 8
  for (int c = 0; c < HIDDEN; ++c)
    q = fmaf(hrow[c], Wq[(size_t)c * ENGRAM + tid], q);
  const float ss = block_reduce_256(q * q, red);
  const float inv = 1.0f / fmaxf(sqrtf(ss), NORM_EPS);
  qn[(size_t)row * ENGRAM + tid] = bf_bits(q * inv);
}

// --------------------------- K3: flash retrieval ---------------------------
// one wave (32 lanes) owns a 16-query tile; streams table in 32-row tiles.
// Both score-B (from tab, k=e contiguous) and accum-B (from tabT, k=t
// contiguous) are single 32-byte contiguous loads per lane -> full pipelining.
__global__ __launch_bounds__(32) void attn_kernel(
    const unsigned short* __restrict__ qn,
    const unsigned short* __restrict__ tab,
    const unsigned short* __restrict__ tabT,
    const float* __restrict__ invn,
    const float* __restrict__ ln_g, const float* __restrict__ ln_b,
    float* __restrict__ retrieved) {
  __shared__ unsigned short ldsP[16 * 32];      // exp(S) tile -> A-matrix src

  const int lane = threadIdx.x;
  const int hf   = lane >> 4;       // lane half (0/1)
  const int nrow = lane & 15;
  const int q0   = blockIdx.x * 16;

  // ---- preload Q A-fragments: 8 chunks of 16x32 bf16
  v16bf qa[8];
  const unsigned short* qrow = qn + (size_t)(q0 + nrow) * ENGRAM;
  #pragma unroll
  for (int c = 0; c < 8; ++c) {
    const unsigned short* p = qrow + c * 32 + hf * 16;
    __builtin_memcpy(&qa[c], p, 32);                                    // k=hf*16+0..15
  }

  v8f acc[16];
  #pragma unroll
  for (int c = 0; c < 16; ++c)
    #pragma unroll
    for (int r = 0; r < 8; ++r) acc[c][r] = 0.0f;
  float rsp[8] = {0, 0, 0, 0, 0, 0, 0, 0};

  for (int t0 = 0; t0 < ENTRIES; t0 += 32) {
    if (t0 + 32 < ENTRIES) {
      __builtin_prefetch(tab + (size_t)(t0 + 32 + lane) * ENGRAM, 0, 0);
      __builtin_prefetch(tabT + (size_t)(lane * 8) * ENTRIES + t0 + 32, 0, 0);
    }

    // ---- scores: two 16x16 tiles, K = 256 in 8 WMMAs each
    v8f s0, s1;
    #pragma unroll
    for (int r = 0; r < 8; ++r) { s0[r] = 0.0f; s1[r] = 0.0f; }
    #pragma unroll
    for (int c = 0; c < 8; ++c) {
      v16bf b0, b1;
      const unsigned short* bp0 =
          tab + (size_t)(t0 + nrow) * ENGRAM + c * 32 + hf * 16;
      const unsigned short* bp1 =
          tab + (size_t)(t0 + 16 + nrow) * ENGRAM + c * 32 + hf * 16;
      __builtin_memcpy(&b0, bp0, 32);   // k = hf*16 + 0..15 (contiguous)
      __builtin_memcpy(&b1, bp1, 32);
      s0 = __builtin_amdgcn_wmma_f32_16x16x32_bf16(false, qa[c], false, b0,
                                                   (short)0, s0, false, false);
      s1 = __builtin_amdgcn_wmma_f32_16x16x32_bf16(false, qa[c], false, b1,
                                                   (short)0, s1, false, false);
    }

    const float inv0 = invn[t0 + nrow];
    const float inv1 = invn[t0 + 16 + nrow];

    // ---- exp (no max needed: |logit| <= 16) + scatter P to LDS
    #pragma unroll
    for (int r = 0; r < 8; ++r) {
      const float p0 = __expf(s0[r] * inv0 * INV_TEMP);
      const float p1 = __expf(s1[r] * inv1 * INV_TEMP);
      rsp[r] += p0 + p1;
      const int m = r + hf * 8;
      ldsP[m * 32 + nrow]      = bf_bits(p0);
      ldsP[m * 32 + 16 + nrow] = bf_bits(p1);
    }
    __syncthreads();

    // ---- P as A-fragment (16x32)
    v16bf pa;
    {
      const unsigned short* pp = &ldsP[nrow * 32 + hf * 16];
      __builtin_memcpy(&pa, pp, 32);
    }

    // ---- retrieved += P(16x32) @ tableTile(32x256): B from transposed table
    #pragma unroll
    for (int c = 0; c < 16; ++c) {
      v16bf vb;
      const unsigned short* vp =
          tabT + (size_t)(c * 16 + nrow) * ENTRIES + t0 + hf * 16;
      __builtin_memcpy(&vb, vp, 32);    // k = hf*16 + 0..15 (contiguous)
      acc[c] = __builtin_amdgcn_wmma_f32_16x16x32_bf16(false, pa, false, vb,
                                                       (short)0, acc[c],
                                                       false, false);
    }
    __syncthreads();  // protect ldsP for next iteration
  }

  // ---- row sums across the 16 lanes sharing each row
  float rsum[8];
  #pragma unroll
  for (int r = 0; r < 8; ++r) {
    float v = rsp[r];
    v += __shfl_xor(v, 1, 16);
    v += __shfl_xor(v, 2, 16);
    v += __shfl_xor(v, 4, 16);
    v += __shfl_xor(v, 8, 16);
    rsum[r] = v;
  }

  // ---- gamma/beta for this lane's 16 columns
  float g16[16], b16[16];
  #pragma unroll
  for (int c = 0; c < 16; ++c) {
    g16[c] = ln_g[c * 16 + nrow];
    b16[c] = ln_b[c * 16 + nrow];
  }

  // ---- normalize by softmax denom + LayerNorm over E, store f32
  #pragma unroll
  for (int r = 0; r < 8; ++r) {
    float s1 = 0.0f, s2 = 0.0f;
    #pragma unroll
    for (int c = 0; c < 16; ++c) {
      const float a = acc[c][r];
      s1 += a; s2 += a * a;
    }
    s1 += __shfl_xor(s1, 1, 16); s2 += __shfl_xor(s2, 1, 16);
    s1 += __shfl_xor(s1, 2, 16); s2 += __shfl_xor(s2, 2, 16);
    s1 += __shfl_xor(s1, 4, 16); s2 += __shfl_xor(s2, 4, 16);
    s1 += __shfl_xor(s1, 8, 16); s2 += __shfl_xor(s2, 8, 16);

    const float inv_rs = 1.0f / rsum[r];
    const float mu  = s1 * inv_rs * (1.0f / 256.0f);
    const float ex2 = s2 * inv_rs * inv_rs * (1.0f / 256.0f);
    const float rstd = rsqrtf(fmaxf(ex2 - mu * mu, 0.0f) + LN_EPS);

    const int m = r + hf * 8;
    float* orow = retrieved + (size_t)(q0 + m) * ENGRAM;
    #pragma unroll
    for (int c = 0; c < 16; ++c) {
      const float x = acc[c][r] * inv_rs;
      orow[c * 16 + nrow] = (x - mu) * rstd * g16[c] + b16[c];
    }
  }
}

// --------------------------- K4: kv proj + gate ----------------------------
__global__ __launch_bounds__(256) void kvgate_kernel(
    const float* __restrict__ hs, const float* __restrict__ retr,
    const float* __restrict__ Wk, const float* __restrict__ Wv,
    const float* __restrict__ gw, const float* __restrict__ kw,
    float* __restrict__ gated, float* __restrict__ gate_out) {
  __shared__ __align__(16) float r[ENGRAM];
  __shared__ float red[256];
  const int row = blockIdx.x, tid = threadIdx.x;
#if HAVE_TDM
  if (tid < 32) {   // wave 0 issues one TDM descriptor: 256 f32 -> LDS
    const unsigned long long ga =
        (unsigned long long)(size_t)(retr + (size_t)row * ENGRAM);
    const unsigned loff = (unsigned)(unsigned long long)(
        __attribute__((address_space(3))) char*)(void*)r;
    v4u_vs g0;
    g0[0] = 1u;                                       // count=1, user desc
    g0[1] = loff;                                     // lds_addr
    g0[2] = (unsigned)(ga & 0xFFFFFFFFu);             // global_addr[31:0]
    g0[3] = (unsigned)((ga >> 32) & 0x1FFFFFFu)       // global_addr[56:32]
            | (2u << 30);                             // type=2 ("image")
    v8i_vs g1;
    g1[0] = 0x20000;            // workgroup_mask=0, data_size=2 (4 bytes)
    g1[1] = (int)(256u << 16);  // tensor_dim0[15:0]=256 (bits 63:48)
    g1[2] = 0;                  // tensor_dim0[31:16]=0, tensor_dim1[15:0]=0
    g1[3] = (int)(256u << 16);  // tile_dim0=256 (bits 127:112)
    g1[4] = 0;                  // tile_dim1=0 (1D tile), tile_dim2=0
    g1[5] = 256;                // tensor_dim0_stride[31:0]
    g1[6] = 0;
    g1[7] = 0;
    v4i_tdm g2 = {0, 0, 0, 0}, g3 = {0, 0, 0, 0};
    v8i_vs  g4 = {0, 0, 0, 0, 0, 0, 0, 0};
    __builtin_amdgcn_tensor_load_to_lds(g0, g1, g2, g3, g4, 0);
    __builtin_amdgcn_s_wait_tensorcnt(0);
  }
  __syncthreads();
#else
  r[tid] = retr[(size_t)row * ENGRAM + tid];
  __syncthreads();
#endif

  float key[8], val[8];
  #pragma unroll
  for (int j = 0; j < 8; ++j) { key[j] = 0.0f; val[j] = 0.0f; }
  for (int e = 0; e < ENGRAM; ++e) {
    const float rv = r[e];
    const float* wkr = Wk + (size_t)e * HIDDEN;
    const float* wvr = Wv + (size_t)e * HIDDEN;
    #pragma unroll
    for (int j = 0; j < 8; ++j) {
      const int c = tid + j * 256;
      key[j] = fmaf(rv, wkr[c], key[j]);
      val[j] = fmaf(rv, wvr[c], val[j]);
    }
  }

  const float* h = hs + (size_t)row * HIDDEN;
  float shh = 0.0f, skk = 0.0f, shk = 0.0f;
  #pragma unroll
  for (int j = 0; j < 8; ++j) {
    const int c = tid + j * 256;
    const float hc = h[c];
    shh = fmaf(hc, hc, shh);
    skk = fmaf(key[j], key[j], skk);
    shk = fmaf(hc * key[j], gw[c] * kw[c], shk);
  }
  shh = block_reduce_256(shh, red);
  skk = block_reduce_256(skk, red);
  shk = block_reduce_256(shk, red);

  const float dot = shk * rsqrtf(shh * (1.0f / HIDDEN) + LN_EPS) *
                          rsqrtf(skk * (1.0f / HIDDEN) + LN_EPS);
  const float raw = 1.0f / (1.0f + __expf(-dot * 0.02209708691207961f)); // /sqrt(2048)
  const float gate = fmaxf(raw, 0.5f);
  #pragma unroll
  for (int j = 0; j < 8; ++j)
    gated[(size_t)row * HIDDEN + tid + j * 256] = gate * val[j];
  if (tid == 0) gate_out[row] = gate;
}

// --------------------------- K5: causal conv + SiLU ------------------------
__global__ __launch_bounds__(256) void convsilu_kernel(
    const float* __restrict__ gated, const float* __restrict__ cw,
    float* __restrict__ out) {
  const size_t idx = (size_t)blockIdx.x * 256 + threadIdx.x;  // < 4096*2048
  const int c = (int)(idx & (HIDDEN - 1));
  const size_t row = idx >> 11;
  const int l = (int)(row & 1023);
  const float w0 = cw[c * 3 + 0], w1 = cw[c * 3 + 1], w2 = cw[c * 3 + 2];
  const float a = (l >= 2) ? gated[idx - 2 * HIDDEN] : 0.0f;
  const float b = (l >= 1) ? gated[idx - HIDDEN] : 0.0f;
  const float g = gated[idx];
  const float conv = w0 * a + w1 * b + w2 * g;
  out[idx] = conv / (1.0f + __expf(-conv));   // SiLU
}

// ---------------------------------------------------------------------------
extern "C" void kernel_launch(void* const* d_in, const int* in_sizes, int n_in,
                              void* d_out, int out_size, void* d_ws,
                              size_t ws_size, hipStream_t stream) {
  (void)in_sizes; (void)n_in; (void)out_size; (void)ws_size;
  const float* hs    = (const float*)d_in[0];
  const float* table = (const float*)d_in[1];
  const float* Wq    = (const float*)d_in[2];
  const float* ln_g  = (const float*)d_in[3];
  const float* ln_b  = (const float*)d_in[4];
  const float* Wk    = (const float*)d_in[5];
  const float* Wv    = (const float*)d_in[6];
  const float* gw    = (const float*)d_in[7];
  const float* kw    = (const float*)d_in[8];
  const float* cw    = (const float*)d_in[9];

  char* ws = (char*)d_ws;
  unsigned short* qn   = (unsigned short*)(ws + QN_OFF);
  unsigned short* tabb = (unsigned short*)(ws + TAB_OFF);
  unsigned short* tabT = (unsigned short*)(ws + TABT_OFF);
  float* invn  = (float*)(ws + INV_OFF);
  float* retr  = (float*)(ws + RET_OFF);
  float* gated = (float*)(ws + GATED_OFF);

  float* out      = (float*)d_out;
  float* gate_out = out + (size_t)ROWS * HIDDEN;   // second tuple output

  tabprep_kernel<<<ENTRIES, 256, 0, stream>>>(table, tabb, invn);
  ttrans_kernel<<<(ENTRIES / 256) * (ENGRAM / 16), 256, 0, stream>>>(tabb,
                                                                     tabT);
  qproj_kernel<<<ROWS, 256, 0, stream>>>(hs, Wq, qn);
  attn_kernel<<<ROWS / 16, 32, 0, stream>>>(qn, tabb, tabT, invn, ln_g, ln_b,
                                            retr);
  kvgate_kernel<<<ROWS, 256, 0, stream>>>(hs, retr, Wk, Wv, gw, kw, gated,
                                          gate_out);
  convsilu_kernel<<<(ROWS * HIDDEN) / 256, 256, 0, stream>>>(gated, cw, out);
}
